// ScaledDotProductAttention_66202625900641
// MI455X (gfx1250) — compile-verified, matrix operand order checked by reading, not run on hardware
//
#include <hip/hip_runtime.h>

typedef __attribute__((ext_vector_type(16))) _Float16 v16h;
typedef __attribute__((ext_vector_type(8)))  _Float16 v8h;
typedef __attribute__((ext_vector_type(8)))  float    v8f;

constexpr int kS     = 2048;
constexpr int kD     = 128;
constexpr int kQTile = 128;      // query rows per block (8 waves x 16 rows)
constexpr int kKT    = 32;       // keys per kv tile (one WMMA K=32 for P*V)
constexpr int kLDK   = kD + 8;   // half-stride of K tile rows   (272B, 16B aligned)
constexpr int kLDV   = kKT + 8;  // half-stride of V^T tile rows (80B, 16B aligned)
constexpr int kLDP   = kKT + 8;  // half-stride of per-wave P tile
constexpr float kScale = 0.08838834764831845f; // 1/sqrt(128)

// v_permlane16_b32-based xor shuffle within each 16-lane half.
// Nibble i of {sel1,sel0} = i ^ m, applied to both halves identically.
__device__ inline float plXor(float v, unsigned s0, unsigned s1) {
  int x = __builtin_bit_cast(int, v);
  int r = __builtin_amdgcn_permlane16(x, x, (int)s0, (int)s1, false, false);
  return __builtin_bit_cast(float, r);
}
__device__ inline float rowMax16(float v) {
  v = fmaxf(v, plXor(v, 0x67452301u, 0xEFCDAB89u)); // xor 1
  v = fmaxf(v, plXor(v, 0x54761032u, 0xDCFE98BAu)); // xor 2
  v = fmaxf(v, plXor(v, 0x32107654u, 0xBA98FEDCu)); // xor 4
  v = fmaxf(v, plXor(v, 0xFEDCBA98u, 0x76543210u)); // xor 8
  return v;
}
__device__ inline float rowSum16(float v) {
  v += plXor(v, 0x67452301u, 0xEFCDAB89u);
  v += plXor(v, 0x54761032u, 0xDCFE98BAu);
  v += plXor(v, 0x32107654u, 0xBA98FEDCu);
  v += plXor(v, 0xFEDCBA98u, 0x76543210u);
  return v;
}

// Pack two fp32 -> two f16 -> one 32-bit LDS store (dst 4B aligned).
__device__ inline void stPair(_Float16* dst, float a, float b) {
  union { _Float16 h[2]; unsigned u; } t;
  t.h[0] = (_Float16)a; t.h[1] = (_Float16)b;
  *(unsigned*)dst = t.u;
}

// Build a 16x32 f16 A/B fragment from an LDS row: lane half (koff) selects
// K 0..7 & 16..23 (lanes 0-15) or K 8..15 & 24..31 (lanes 16-31).
__device__ inline v16h fragLds(const _Float16* base, int koff) {
  v8h a = *(const v8h*)(base + koff);
  v8h b = *(const v8h*)(base + 16 + koff);
  v16h f;
#pragma unroll
  for (int i = 0; i < 8; ++i) { f[i] = a[i]; f[i + 8] = b[i]; }
  return f;
}

__global__ __launch_bounds__(256, 2)
void fa_fwd_causal(const float* __restrict__ Q, const float* __restrict__ K,
                   const float* __restrict__ V, float* __restrict__ O) {
  __shared__ __align__(16) _Float16 sk [kKT * kLDK];      // K tile, row-major
  __shared__ __align__(16) _Float16 svt[kD  * kLDV];      // V tile, transposed
  __shared__ __align__(16) _Float16 sp [8 * 16 * kLDP];   // per-wave P tiles

  const int tid  = threadIdx.x;
  const int wave = tid >> 5;
  const int lane = tid & 31;
  const int ln   = lane & 15;
  const int hi   = lane >> 4;
  const int koff = hi * 8;

  const int bh  = blockIdx.y;
  const int q0b = blockIdx.x * kQTile;
  const int q0w = q0b + wave * 16;
  // Wave-uniform copy so the causal skip compiles to a scalar branch
  // (keeps EXEC untouched around the WMMA region).
  const int q0wS = __builtin_amdgcn_readfirstlane(q0w);

  const float* qp = Q + (size_t)bh * kS * kD;
  const float* kp = K + (size_t)bh * kS * kD;
  const float* vp = V + (size_t)bh * kS * kD;
  float*       op = O + (size_t)bh * kS * kD;

  // Load + pre-scale this wave's 16x128 Q tile into 4 f16 A-fragments.
  const float* qrow = qp + (size_t)(q0w + ln) * kD;
  v16h qf[4];
#pragma unroll
  for (int c = 0; c < 4; ++c) {
    const float* p0 = qrow + 32 * c + koff;
#pragma unroll
    for (int i = 0; i < 8; ++i) {
      qf[c][i]     = (_Float16)(p0[i] * kScale);
      qf[c][i + 8] = (_Float16)(p0[16 + i] * kScale);
    }
  }

  v8f acc[8] = {};
  float m[8], l[8];
#pragma unroll
  for (int i = 0; i < 8; ++i) { m[i] = -__builtin_inff(); l[i] = 0.0f; }

  _Float16* pw = sp + wave * 16 * kLDP;
  const int kvEnd = q0b + kQTile;

  for (int kv = 0; kv < kvEnd; kv += kKT) {
    // Cooperative stage in 2x2 (key x d) patches:
    // 4 coalesced float2 global loads -> 4 packed b32 LDS stores.
#pragma unroll
    for (int it = 0; it < (kKT / 2) * (kD / 2) / 256; ++it) {
      const int p    = it * 256 + tid;
      const int key  = (p >> 6) << 1;          // 64 d-pairs per key pair
      const int d    = (p & 63) << 1;
      const float2 k0 = *(const float2*)(kp + (size_t)(kv + key)     * kD + d);
      const float2 k1 = *(const float2*)(kp + (size_t)(kv + key + 1) * kD + d);
      const float2 v0 = *(const float2*)(vp + (size_t)(kv + key)     * kD + d);
      const float2 v1 = *(const float2*)(vp + (size_t)(kv + key + 1) * kD + d);
      stPair(sk + (key)     * kLDK + d, k0.x, k0.y);
      stPair(sk + (key + 1) * kLDK + d, k1.x, k1.y);
      stPair(svt + (d)     * kLDV + key, v0.x, v1.x);
      stPair(svt + (d + 1) * kLDV + key, v0.y, v1.y);
    }
    // Prefetch next KV tile into cache while this tile's loads drain.
    if (kv + kKT < kvEnd) {
      const int r    = tid >> 3;               // 32 rows
      const int cseg = (tid & 7) * 16;         // 8 x 64B segments per row
      const size_t off = (size_t)(kv + kKT + r) * kD + cseg;
      __builtin_prefetch(kp + off, 0, 1);
      __builtin_prefetch(vp + off, 0, 1);
    }
    __syncthreads();

    if (kv <= q0wS + 15) {               // scalar, wave-uniform causal skip
      const bool needMask = (kv + kKT - 1) > q0wS;
      // S = Q*K^T for both 16-key subtiles (8 WMMAs).
      v8f s0 = {}, s1 = {};
#pragma unroll
      for (int c = 0; c < 4; ++c) {
        v16h bf0 = fragLds(sk + (ln)      * kLDK + 32 * c, koff);
        v16h bf1 = fragLds(sk + (16 + ln) * kLDK + 32 * c, koff);
        s0 = __builtin_amdgcn_wmma_f32_16x16x32_f16(false, qf[c], false, bf0,
                                                    (short)0, s0, false, false);
        s1 = __builtin_amdgcn_wmma_f32_16x16x32_f16(false, qf[c], false, bf1,
                                                    (short)0, s1, false, false);
      }
      // One combined online-softmax update for the whole 32-key tile.
      const int col0 = kv + ln;
      const int col1 = kv + 16 + ln;
#pragma unroll
      for (int i = 0; i < 8; ++i) {
        const int rowg = q0w + i + 8 * hi;
        float a0 = s0[i], a1 = s1[i];
        if (needMask && col0 > rowg) a0 = -__builtin_inff();
        if (needMask && col1 > rowg) a1 = -__builtin_inff();
        const float mt   = rowMax16(fmaxf(a0, a1));
        const float mnew = fmaxf(m[i], mt);
        const float al   = __expf(m[i] - mnew);
        const float p0   = __expf(a0 - mnew);
        const float p1   = __expf(a1 - mnew);
        l[i] = l[i] * al + rowSum16(p0 + p1);
        m[i] = mnew;
#pragma unroll
        for (int n = 0; n < 8; ++n) acc[n][i] *= al;
        pw[(i + 8 * hi) * kLDP + ln]      = (_Float16)p0;   // C->A relayout
        pw[(i + 8 * hi) * kLDP + 16 + ln] = (_Float16)p1;
      }
      // P (16x32) x V (32x128): one A-fragment, 8 accumulating WMMAs.
      v16h pa = fragLds(pw + ln * kLDP, koff);
#pragma unroll
      for (int n = 0; n < 8; ++n) {
        v16h vb = fragLds(svt + (n * 16 + ln) * kLDV, koff);
        acc[n] = __builtin_amdgcn_wmma_f32_16x16x32_f16(false, pa, false, vb,
                                                        (short)0, acc[n], false, false);
      }
    }
    __syncthreads();
  }

  // Normalize by running sum and store fp32 (coalesced b32 stores).
#pragma unroll
  for (int i = 0; i < 8; ++i) {
    const float inv = 1.0f / l[i];
    float* orow = op + (size_t)(q0w + i + 8 * hi) * kD;
#pragma unroll
    for (int n = 0; n < 8; ++n) orow[n * 16 + ln] = acc[n][i] * inv;
  }
}

extern "C" void kernel_launch(void* const* d_in, const int* in_sizes, int n_in,
                              void* d_out, int out_size, void* d_ws, size_t ws_size,
                              hipStream_t stream) {
  const float* q = (const float*)d_in[0];
  const float* k = (const float*)d_in[1];
  const float* v = (const float*)d_in[2];
  float* o = (float*)d_out;
  (void)in_sizes; (void)n_in; (void)out_size; (void)d_ws; (void)ws_size;
  dim3 grid(kS / kQTile, 2 * 16);  // (S/128 query tiles, B*H heads)
  dim3 block(256);                 // 8 wave32s
  fa_fwd_causal<<<grid, block, 0, stream>>>(q, k, v, o);
}
